// FastWeightMemory_71416716197986
// MI455X (gfx1250) — matrix operand order repaired and checked by reading, not compiled, not run on hardware
//
#include <hip/hip_runtime.h>
#include <hip/hip_bf16.h>

// ---------------------------------------------------------------------------
// FastWeightMemory: out = softmax_n( (k_norm . q_norm)/0.1 )^T @ V
// SIZE(N)=32768 keys, DIM(D)=1024, BATCH(B)=4096, softmax over N.
// Fused flash-attention-style kernel using bf16 WMMA on gfx1250 (wave32).
// ---------------------------------------------------------------------------

#define N_KEYS 32768
#define DIM    1024
#define BATCH  4096
#define BT     32          // queries per workgroup (2 WMMA M-tiles)
#define KT     32          // keys per inner tile (WMMA K for PV, 2 N-tiles for QK)
#define NWAVE  8           // waves per workgroup
#define KSLICE 128         // dim-columns owned by each wave (QK reduction / PV output)

typedef __attribute__((ext_vector_type(16))) __bf16 v16bf;
typedef __attribute__((ext_vector_type(8)))  __bf16 v8bf;
typedef __attribute__((ext_vector_type(8)))  float  v8f;
typedef __attribute__((ext_vector_type(4)))  float  v4f;

static __device__ __forceinline__ v16bf cat16(v8bf lo, v8bf hi) {
  return __builtin_shufflevector(lo, hi, 0,1,2,3,4,5,6,7,8,9,10,11,12,13,14,15);
}

static __device__ __forceinline__ v8f wmma_bf16(v16bf a, v16bf b, v8f c) {
  // D = A(16x32 bf16) * B(32x16 bf16) + C(16x16 f32)
  return __builtin_amdgcn_wmma_f32_16x16x32_bf16(false, a, false, b, (short)0, c,
                                                 false, false);
}

// A-operand (16x32, row-major source, row stride `strideElems` elements):
// lane: m = lane&15, h = lane>>4; elems 0-7 at k0+8h, elems 8-15 at k0+16+8h.
static __device__ __forceinline__ v16bf load_A(const __bf16* base, int row,
                                               int k0, int strideElems,
                                               int m16, int h) {
  const __bf16* p = base + (size_t)row * strideElems + k0 + 8 * h;
  v8bf lo = *(const v8bf*)p;
  v8bf hi = *(const v8bf*)(p + 16);
  return cat16(lo, hi);
}

// B-operand (32x16): lane: n = lane&15, h = lane>>4; elems 0-7 at k0+8h,
// elems 8-15 at k0+16+8h, from source row n (source row-major over K).
static __device__ __forceinline__ v16bf load_B(const __bf16* base, int row,
                                               int k0, int strideElems, int h) {
  const __bf16* p = base + (size_t)row * strideElems + k0 + 8 * h;
  v8bf lo = *(const v8bf*)p;
  v8bf hi = *(const v8bf*)(p + 16);
  return cat16(lo, hi);
}

// ---------------------------------------------------------------------------
// Preprocess 1: L2-normalize rows (scale folds in 1/TEMP for queries), f32->bf16
// ---------------------------------------------------------------------------
__global__ void rownorm_bf16(const float* __restrict__ src,
                             __bf16* __restrict__ dst, float scale) {
  const int row = blockIdx.x;
  const int t   = threadIdx.x;              // 256 threads
  const float* s = src + (size_t)row * DIM;

  float v[4];
  float acc = 0.f;
#pragma unroll
  for (int i = 0; i < 4; ++i) { v[i] = s[t + 256 * i]; acc += v[i] * v[i]; }
#pragma unroll
  for (int m = 16; m >= 1; m >>= 1) acc += __shfl_xor(acc, m, 32);

  __shared__ float red[8];
  if ((t & 31) == 0) red[t >> 5] = acc;
  __syncthreads();
  float tot = 0.f;
#pragma unroll
  for (int w = 0; w < 8; ++w) tot += red[w];

  const float r = scale / fmaxf(sqrtf(tot), 1e-12f);
  __bf16* d = dst + (size_t)row * DIM;
#pragma unroll
  for (int i = 0; i < 4; ++i) d[t + 256 * i] = (__bf16)(v[i] * r);
}

// ---------------------------------------------------------------------------
// Preprocess 2: V [N,D] f32  ->  Vt [D,N] bf16 (LDS-tiled transpose)
// ---------------------------------------------------------------------------
__global__ void transpose_bf16(const float* __restrict__ v,
                               __bf16* __restrict__ vt) {
  __shared__ __bf16 tile[32][33];
  const int nb = blockIdx.x * 32, db = blockIdx.y * 32;
  const int tx = threadIdx.x & 31, ty = threadIdx.x >> 5;   // 32x8
#pragma unroll
  for (int i = 0; i < 32; i += 8)
    tile[ty + i][tx] = (__bf16)v[(size_t)(nb + ty + i) * DIM + db + tx];
  __syncthreads();
#pragma unroll
  for (int i = 0; i < 32; i += 8)
    vt[(size_t)(db + ty + i) * N_KEYS + nb + tx] = tile[tx][ty + i];
}

// ---------------------------------------------------------------------------
// Main fused kernel. 8 waves, 32 queries per WG.
//  P1: each wave: partial S(32x32) over its 128-dim slice  (16 WMMA bf16)
//  P2: partial S -> LDS (b128 stores)
//  P3: waves 0/1: reduce partials, online softmax, write alpha + bf16 P
//  P4: all waves: rescale O, P@V over own 128 output columns (16 WMMA bf16)
// ---------------------------------------------------------------------------
__global__ __launch_bounds__(256, 1) void fwm_attn(
    const __bf16* __restrict__ qn,   // [BATCH][DIM]  (pre-scaled by 1/TEMP)
    const __bf16* __restrict__ kn,   // [N_KEYS][DIM]
    const __bf16* __restrict__ vt,   // [DIM][N_KEYS]
    float* __restrict__ out)         // [BATCH][DIM] f32
{
  __shared__ float  lds_S[NWAVE][8][32][4];   // [wave][r][lane][tile] 32 KB
  __shared__ __bf16 lds_P[BT][KT];            // probs, row-major       2 KB
  __shared__ float  lds_a[BT];                // per-row rescale alpha
  __shared__ float  lds_l[BT];                // per-row final denom

  const int tid  = threadIdx.x;
  const int wave = tid >> 5;
  const int lane = tid & 31;
  const int m16  = lane & 15;
  const int h    = lane >> 4;
  const int qb   = blockIdx.x * BT;
  const int ksl  = wave * KSLICE;   // this wave's K-dim slice (QK reduction)
  const int csl  = wave * KSLICE;   // this wave's output-column slice (PV)

  // ---- preload Q A-operands: [mtile][kchunk], resident for whole kernel ----
  v16bf QA[2][4];
#pragma unroll
  for (int mt = 0; mt < 2; ++mt)
#pragma unroll
    for (int kc = 0; kc < 4; ++kc)
      QA[mt][kc] = load_A(qn, qb + mt * 16 + m16, ksl + kc * 32, DIM, m16, h);

  // ---- O accumulators (C-layout: row = r + 8h, col = csl + ct*16 + m16) ----
  v8f O[2][8];
#pragma unroll
  for (int mt = 0; mt < 2; ++mt)
#pragma unroll
    for (int ct = 0; ct < 8; ++ct) O[mt][ct] = (v8f)0.f;

  // online-softmax state, owned by wave `mt` (per lane: rows r+8h of its mtile)
  float mrow[8], lrow[8];
#pragma unroll
  for (int r = 0; r < 8; ++r) { mrow[r] = -1e30f; lrow[r] = 0.f; }

  for (int kt = 0; kt < N_KEYS / KT; ++kt) {
    const int keyBase = kt * KT;

    // speculative prefetch of the tile after next (global_prefetch_b8)
    if (kt + 2 < N_KEYS / KT) {
      __builtin_prefetch(kn + (size_t)(keyBase + 2 * KT + lane) * DIM + ksl, 0, 1);
      __builtin_prefetch(vt + (size_t)(csl + lane) * N_KEYS + keyBase + 2 * KT, 0, 1);
      __builtin_prefetch(vt + (size_t)(csl + 64 + lane) * N_KEYS + keyBase + 2 * KT, 0, 1);
    }

    // ---- P1: partial S over this wave's 128-dim slice ----
    v8f S[2][2];
#pragma unroll
    for (int mt = 0; mt < 2; ++mt)
#pragma unroll
      for (int nt = 0; nt < 2; ++nt) S[mt][nt] = (v8f)0.f;

#pragma unroll
    for (int nt = 0; nt < 2; ++nt) {
#pragma unroll
      for (int kc = 0; kc < 4; ++kc) {
        v16bf KB = load_B(kn, keyBase + nt * 16 + m16, ksl + kc * 32, DIM, h);
        S[0][nt] = wmma_bf16(QA[0][kc], KB, S[0][nt]);
        S[1][nt] = wmma_bf16(QA[1][kc], KB, S[1][nt]);
      }
    }

    // ---- P2: spill partial S to LDS (one b128 per r) ----
#pragma unroll
    for (int r = 0; r < 8; ++r) {
      v4f pk = {S[0][0][r], S[0][1][r], S[1][0][r], S[1][1][r]};
      *(v4f*)&lds_S[wave][r][lane][0] = pk;
    }
    __syncthreads();

    // ---- P3: waves 0/1 reduce + online softmax for mtile == wave ----
    if (wave < 2) {
      const int mt = wave;
      float s2[2][8];
#pragma unroll
      for (int r = 0; r < 8; ++r) {
        v4f acc = (v4f)0.f;
#pragma unroll
        for (int w = 0; w < NWAVE; ++w)
          acc += *(const v4f*)&lds_S[w][r][lane][0];
        s2[0][r] = acc[2 * mt + 0];
        s2[1][r] = acc[2 * mt + 1];
      }
#pragma unroll
      for (int r = 0; r < 8; ++r) {
        const int row = mt * 16 + r + 8 * h;
        float rm = fmaxf(s2[0][r], s2[1][r]);
#pragma unroll
        for (int mk = 8; mk >= 1; mk >>= 1) rm = fmaxf(rm, __shfl_xor(rm, mk, 32));
        const float mnew  = fmaxf(mrow[r], rm);
        const float alpha = __expf(mrow[r] - mnew);
        mrow[r] = mnew;
        const float p0 = __expf(s2[0][r] - mnew);
        const float p1 = __expf(s2[1][r] - mnew);
        float rs = p0 + p1;
#pragma unroll
        for (int mk = 8; mk >= 1; mk >>= 1) rs += __shfl_xor(rs, mk, 32);
        lrow[r] = lrow[r] * alpha + rs;
        if (m16 == 0) lds_a[row] = alpha;
        lds_P[row][m16]      = (__bf16)p0;
        lds_P[row][16 + m16] = (__bf16)p1;
      }
    }
    __syncthreads();

    // ---- P4: rescale O, accumulate P @ V on own 128 columns ----
    float a2[2][8];
#pragma unroll
    for (int mt = 0; mt < 2; ++mt)
#pragma unroll
      for (int r = 0; r < 8; ++r) a2[mt][r] = lds_a[mt * 16 + r + 8 * h];
#pragma unroll
    for (int mt = 0; mt < 2; ++mt)
#pragma unroll
      for (int ct = 0; ct < 8; ++ct)
#pragma unroll
        for (int r = 0; r < 8; ++r) O[mt][ct][r] *= a2[mt][r];

    v16bf PA[2];
#pragma unroll
    for (int mt = 0; mt < 2; ++mt)
      PA[mt] = load_A(&lds_P[0][0], mt * 16 + m16, 0, KT, m16, h);

#pragma unroll
    for (int ct = 0; ct < 8; ++ct) {
      v16bf VB = load_B(vt, csl + ct * 16 + m16, keyBase, N_KEYS, h);
      O[0][ct] = wmma_bf16(PA[0], VB, O[0][ct]);
      O[1][ct] = wmma_bf16(PA[1], VB, O[1][ct]);
    }
  }

  // ---- epilogue: publish denominators, normalize, store f32 ----
  if (wave < 2 && m16 == 0) {
    const int mt = wave;
#pragma unroll
    for (int r = 0; r < 8; ++r) lds_l[mt * 16 + r + 8 * h] = lrow[r];
  }
  __syncthreads();

#pragma unroll
  for (int mt = 0; mt < 2; ++mt)
#pragma unroll
    for (int r = 0; r < 8; ++r) {
      const int row = mt * 16 + r + 8 * h;
      const float inv = 1.f / lds_l[row];
      const int b = qb + row;
#pragma unroll
      for (int ct = 0; ct < 8; ++ct)
        out[(size_t)b * DIM + csl + ct * 16 + m16] = O[mt][ct][r] * inv;
    }
}

// ---------------------------------------------------------------------------
extern "C" void kernel_launch(void* const* d_in, const int* in_sizes, int n_in,
                              void* d_out, int out_size, void* d_ws, size_t ws_size,
                              hipStream_t stream) {
  const float* query  = (const float*)d_in[0];   // [4096][1024]
  const float* keys   = (const float*)d_in[1];   // [32768][1024]
  const float* values = (const float*)d_in[2];   // [32768][1024]
  float* out = (float*)d_out;                    // [4096][1024]

  // workspace: qn (8 MB) | kn (64 MB) | vt (64 MB), all bf16
  __bf16* qn = (__bf16*)d_ws;
  __bf16* kn = qn + (size_t)BATCH * DIM;
  __bf16* vt = kn + (size_t)N_KEYS * DIM;

  rownorm_bf16<<<BATCH, 256, 0, stream>>>(query, qn, 10.0f);   // fold 1/TEMP
  rownorm_bf16<<<N_KEYS, 256, 0, stream>>>(keys, kn, 1.0f);
  transpose_bf16<<<dim3(N_KEYS / 32, DIM / 32), 256, 0, stream>>>(values, vt);
  fwm_attn<<<BATCH / BT, 32 * NWAVE, 0, stream>>>(qn, kn, vt, out);
}
